// RNN_57174604644504
// MI455X (gfx1250) — matrix operand order repaired and checked by reading, not compiled, not run on hardware
//
#include <hip/hip_runtime.h>
#include <cstddef>
#include <cstdint>

// ---------------------------------------------------------------------------
// RNN: Xw = X@W^T (fused), h_t = tanh(Xw_t + h@U^T + U_b) scan, Y = hs@V^T + V_b
// B=256, T=512, I=128, H=1024, O=64.  Matmuls via v_wmma_f32_16x16x32_bf16;
// tile staging via TDM tensor_load_to_lds (TENSORcnt).
// ---------------------------------------------------------------------------

namespace {
constexpr int Bb = 256, Tt = 512, Ii = 128, Hh = 1024, Oo = 64;
constexpr int GROUPS = 16;   // batch groups of 16 rows
constexpr int SLICES = 16;   // H slices of 64 features
constexpr int ROWS   = 16;   // batch rows per WG (WMMA M)
constexpr int FSL    = 64;   // h-features per WG (4 waves x N=16)

// workspace layout (bytes)
constexpr size_t OFF_UBF = 0;              // U as bf16, [H][H] row-major             2 MB
constexpr size_t OFF_HB0 = 4u << 20;       // h0 as bf16 [B][H]                       512 KB
constexpr size_t OFF_VBF = 5u << 20;       // V as bf16 [O][H]                        128 KB
constexpr size_t OFF_CNT = 6u << 20;       // 16 group barrier counters
constexpr size_t OFF_HS  = 8u << 20;       // hs as bf16 [B][T][H]                    256 MB
}

typedef __attribute__((ext_vector_type(16))) __bf16 v16bf;
typedef __attribute__((ext_vector_type(8)))  float  v8f;
typedef __attribute__((ext_vector_type(4)))  unsigned int u32x4;
typedef __attribute__((ext_vector_type(8)))  int i32x8;
typedef __attribute__((ext_vector_type(4)))  int i32x4;

union FragBF { v16bf v; uint4 q[2]; };

__device__ __forceinline__ unsigned short f2bf(float f) {
  unsigned int u = __float_as_uint(f);
  unsigned int r = u + 0x7FFFu + ((u >> 16) & 1u);   // RNE
  return (unsigned short)(r >> 16);
}

__device__ __forceinline__ v8f wmma_bf16(const FragBF& a, const FragBF& b, v8f c) {
  return __builtin_amdgcn_wmma_f32_16x16x32_bf16(false, a.v, false, b.v,
                                                 (short)0, c, false, false);
}

// A fragment: per-lane row M=lane&15; K = kb + (e&7) + 8*hi + 16*(e>=8)
__device__ __forceinline__ void loadA(FragBF& f, const unsigned short* rowbase,
                                      int kb, int hi) {
  const uint4* p = reinterpret_cast<const uint4*>(rowbase + kb + hi * 8);
  f.q[0] = p[0];
  f.q[1] = p[2];   // +16 elements
}
// B fragment: per-lane col N=lane&15; K = kb + (e&7) + 8*hi + 16*(e>=8)
__device__ __forceinline__ void loadB(FragBF& f, const unsigned short* colbase,
                                      int kb, int hi) {
  const uint4* p = reinterpret_cast<const uint4*>(colbase + kb + hi * 8);
  f.q[0] = p[0];
  f.q[1] = p[2];
}

// ---------------------------------------------------------------------------
// TDM: DMA a 2D tile of bf16 (tile_d1 rows x tile_d0 elems, row pitch
// row_stride elems) from global memory into LDS at lds_off.  One wave issues;
// completion tracked with TENSORcnt.  D# layout per CDNA5 ISA ch.8.
// ---------------------------------------------------------------------------
__device__ __forceinline__ void tdm_load_2d(unsigned lds_off, const void* gsrc,
                                            unsigned tile_d0, unsigned tile_d1,
                                            unsigned long long row_stride) {
  unsigned long long ga = (unsigned long long)(uintptr_t)gsrc;
  const unsigned td0 = 0x7FFFFFFFu;   // tensor dims huge -> no OOB clipping
  const unsigned td1 = 0x7FFFFFFFu;
  u32x4 g0;
  g0[0] = 1u;                                           // count=1, user D#
  g0[1] = lds_off;                                      // lds_addr (bytes)
  g0[2] = (unsigned)(ga & 0xFFFFFFFFu);                 // global_addr lo
  g0[3] = (unsigned)((ga >> 32) & 0x01FFFFFFu)          // global_addr hi
        | (2u << 30);                                   // type = 2 (image)
  i32x8 g1;
  g1[0] = (int)(1u << 16);                              // data_size=1 (2 bytes)
  g1[1] = (int)((td0 & 0xFFFFu) << 16);                 // tensor_dim0 lo16
  g1[2] = (int)((td0 >> 16) | ((td1 & 0xFFFFu) << 16)); // dim0 hi / dim1 lo
  g1[3] = (int)((td1 >> 16) | (tile_d0 << 16));         // dim1 hi / tile_dim0
  g1[4] = (int)(tile_d1 & 0xFFFFu);                     // tile_dim1, tile_dim2=0
  g1[5] = (int)(unsigned)(row_stride & 0xFFFFFFFFull);  // tensor_dim0_stride lo
  g1[6] = (int)(unsigned)((row_stride >> 32) & 0xFFFFull); // stride hi, d1s=0
  g1[7] = 0;
  i32x4 z4 = {0, 0, 0, 0};                              // groups 2/3: <=2D tensor
  i32x8 z8 = {0, 0, 0, 0, 0, 0, 0, 0};
  __builtin_amdgcn_tensor_load_to_lds(g0, g1, z4, z4, z8, 0);
}

// ---------------------------------------------------------------------------
// Init: convert U, h0, V to bf16 in ws; zero group barrier counters.
// ---------------------------------------------------------------------------
__global__ void rnn_init(const float* __restrict__ h0, const float* __restrict__ U,
                         const float* __restrict__ V,
                         unsigned short* __restrict__ Ubf,
                         unsigned short* __restrict__ hb0,
                         unsigned short* __restrict__ Vbf,
                         unsigned int* __restrict__ cnt) {
  size_t idx = (size_t)blockIdx.x * blockDim.x + threadIdx.x;
  size_t stride = (size_t)gridDim.x * blockDim.x;
  for (size_t i = idx; i < (size_t)Hh * Hh; i += stride) Ubf[i] = f2bf(U[i]);
  for (size_t i = idx; i < (size_t)Bb * Hh; i += stride) hb0[i] = f2bf(h0[i]);
  for (size_t i = idx; i < (size_t)Oo * Hh; i += stride) Vbf[i] = f2bf(V[i]);
  if (idx < GROUPS) cnt[idx] = 0u;
}

// ---------------------------------------------------------------------------
// Persistent scan: 256 WGs = 16 batch-groups x 16 H-slices, 4 waves each.
// Per step per wave: 4 WMMA (X·W^T, W-slice in LDS) + 32 WMMA (h·U^T, U from L2),
// bias + tanh, bf16 h_new -> hs.  h_old tile staged by TDM; per-group
// global-atomic barrier between steps.
// ---------------------------------------------------------------------------
__global__ void __launch_bounds__(128)
rnn_scan(const float* __restrict__ X, const float* __restrict__ W,
         const float* __restrict__ U_b,
         const unsigned short* __restrict__ Ubf,
         const unsigned short* __restrict__ hb0,
         unsigned short* __restrict__ hs,
         unsigned int* __restrict__ cnt,
         float* __restrict__ Hlast) {
  __shared__ __align__(16) unsigned short Hs[ROWS * Hh];   // 32 KB  h_old tile (bf16)
  __shared__ __align__(16) unsigned short Xs[ROWS * Ii];   //  4 KB  X tile (bf16)
  __shared__ __align__(16) unsigned short Wsl[FSL * Ii];   // 16 KB  W slice, [n][k]

  const int tid  = threadIdx.x;
  const int lane = tid & 31;
  const int ln   = lane & 15;
  const int hi   = lane >> 4;
  const int wave = tid >> 5;
  const int n0   = wave * 16;

  const int g  = blockIdx.x >> 4;   // batch group
  const int s  = blockIdx.x & 15;   // H slice
  const int R0 = g * ROWS;
  const int F0 = s * FSL;

  // stage W slice into LDS as bf16, [n][k] (k contiguous)
  for (int i = tid; i < FSL * Ii; i += 128) {
    int n = i >> 7, k = i & (Ii - 1);
    Wsl[i] = f2bf(W[(size_t)(F0 + n) * Ii + k]);
  }
  const float ub = U_b[F0 + n0 + ln];   // bias, per output column

  __syncthreads();

  float hval[8];
  for (int r = 0; r < 8; ++r) hval[r] = 0.f;

#pragma unroll 1
  for (int t = 0; t < Tt; ++t) {
    // ---- TDM: stage h_old tile (16 rows x 1024 bf16) into LDS
    if (tid < 32) {
      const unsigned short* src =
          (t == 0) ? hb0 + (size_t)R0 * Hh
                   : hs + ((size_t)R0 * Tt + (t - 1)) * Hh;
      unsigned long long pitch = (t == 0) ? (unsigned long long)Hh
                                          : (unsigned long long)Tt * Hh;
      tdm_load_2d((unsigned)(uintptr_t)Hs, src, Hh, ROWS, pitch);
    }
    // ---- stage X[:, t, :] tile, fp32 -> bf16 (overlaps with TDM)
    {
      int m = tid >> 3, seg = tid & 7;
      const float2* xr = reinterpret_cast<const float2*>(
          X + ((size_t)(R0 + m) * Tt + t) * Ii + seg * 16);
      unsigned int* xd =
          reinterpret_cast<unsigned int*>(Xs) + ((m * Ii + seg * 16) >> 1);
#pragma unroll
      for (int j = 0; j < 8; ++j) {
        float2 f2 = xr[j];
        xd[j] = (unsigned int)f2bf(f2.x) | ((unsigned int)f2bf(f2.y) << 16);
      }
    }
    // prefetch next timestep's X rows (global_prefetch_b8)
    if (t + 1 < Tt && tid < ROWS) {
      __builtin_prefetch(X + ((size_t)(R0 + tid) * Tt + (t + 1)) * Ii, 0, 0);
    }
    if (tid < 32) {
      __builtin_amdgcn_s_wait_tensorcnt(0);   // h_old tile landed in LDS
    }
    __syncthreads();

    v8f acc;
#pragma unroll
    for (int r = 0; r < 8; ++r) acc[r] = 0.f;

    // ---- fused input projection: acc += X_tile(16x128) * W_slice^T  (4 WMMA)
    {
      const unsigned short* ax = Xs + (unsigned)ln * Ii;
      const unsigned short* bw = Wsl + (unsigned)(n0 + ln) * Ii;
#pragma unroll
      for (int kb = 0; kb < Ii; kb += 32) {
        FragBF a, b;
        loadA(a, ax, kb, hi);
        loadB(b, bw, kb, hi);
        acc = wmma_bf16(a, b, acc);
      }
    }
    // ---- recurrence: acc += h_old(16x1024) * U_slice^T  (32 WMMA, U from L2)
    {
      const unsigned short* ah = Hs + (unsigned)ln * Hh;
      const unsigned short* bu = Ubf + (size_t)(F0 + n0 + ln) * Hh;
#pragma unroll 4
      for (int kb = 0; kb < Hh; kb += 32) {
        FragBF a, b;
        loadA(a, ah, kb, hi);
        loadB(b, bu, kb, hi);
        acc = wmma_bf16(a, b, acc);
      }
    }

    // ---- bias + tanh, write h_new (bf16) into hs[b][t][h]
#pragma unroll
    for (int r = 0; r < 8; ++r) {
      float h = tanhf(acc[r] + ub);
      hval[r] = h;
      int m = r + hi * 8;
      hs[((size_t)(R0 + m) * Tt + t) * Hh + F0 + n0 + ln] = f2bf(h);
    }

    // ---- per-group barrier: make all 16 slices' h_new visible before t+1
    if (t + 1 < Tt) {
      __threadfence();          // release this wave's hs stores (device scope)
      __syncthreads();          // all waves of the WG released
      if (tid == 0) {
        __hip_atomic_fetch_add(&cnt[g], 1u, __ATOMIC_RELEASE,
                               __HIP_MEMORY_SCOPE_AGENT);
        unsigned target = (unsigned)(SLICES * (t + 1));
        while (__hip_atomic_load(&cnt[g], __ATOMIC_ACQUIRE,
                                 __HIP_MEMORY_SCOPE_AGENT) < target) {
          __builtin_amdgcn_s_sleep(8);
        }
      }
      __syncthreads();
      __threadfence();          // acquire: invalidate stale cached h lines
    }
  }

  // ---- h_last (fp32) -> d_out tail
#pragma unroll
  for (int r = 0; r < 8; ++r) {
    int m = r + hi * 8;
    Hlast[(size_t)(R0 + m) * Hh + F0 + n0 + ln] = hval[r];
  }
}

// ---------------------------------------------------------------------------
// Output GEMM: Y[b,t,o] = hs[b,t,:] . V[o,:] + V_b[o]
// M = B*T = 131072 rows, N = 64, K = 1024.  8192 WGs x 4 waves, 32 WMMA each.
// A tile staged by TDM.
// ---------------------------------------------------------------------------
__global__ void __launch_bounds__(128)
rnn_out(const unsigned short* __restrict__ hs,
        const unsigned short* __restrict__ Vbf,
        const float* __restrict__ V_b, float* __restrict__ Y) {
  __shared__ __align__(16) unsigned short Hs2[ROWS * Hh];   // 32 KB

  const int tid  = threadIdx.x;
  const int lane = tid & 31;
  const int ln   = lane & 15;
  const int hi   = lane >> 4;
  const int wave = tid >> 5;
  const int RT0  = blockIdx.x * ROWS;          // flat (b*T + t) row tile
  const int o    = wave * 16 + ln;             // output column

  if (tid < 32) {
    tdm_load_2d((unsigned)(uintptr_t)Hs2, hs + (size_t)RT0 * Hh, Hh, ROWS,
                (unsigned long long)Hh);
    __builtin_amdgcn_s_wait_tensorcnt(0);
  }
  const float vb = V_b[o];
  __syncthreads();

  v8f acc;
#pragma unroll
  for (int r = 0; r < 8; ++r) acc[r] = 0.f;

  const unsigned short* ah = Hs2 + (unsigned)ln * Hh;
  const unsigned short* bv = Vbf + (size_t)o * Hh;
#pragma unroll 4
  for (int kb = 0; kb < Hh; kb += 32) {
    FragBF a, b;
    loadA(a, ah, kb, hi);
    loadB(b, bv, kb, hi);
    acc = wmma_bf16(a, b, acc);
  }

#pragma unroll
  for (int r = 0; r < 8; ++r) {
    int m = r + hi * 8;
    Y[(size_t)(RT0 + m) * Oo + o] = acc[r] + vb;
  }
}

// ---------------------------------------------------------------------------
extern "C" void kernel_launch(void* const* d_in, const int* in_sizes, int n_in,
                              void* d_out, int out_size, void* d_ws, size_t ws_size,
                              hipStream_t stream) {
  (void)in_sizes; (void)n_in; (void)out_size; (void)ws_size;
  const float* X   = (const float*)d_in[0];
  const float* h0  = (const float*)d_in[1];
  const float* W   = (const float*)d_in[2];
  const float* U   = (const float*)d_in[3];
  const float* U_b = (const float*)d_in[4];
  const float* V   = (const float*)d_in[5];
  const float* V_b = (const float*)d_in[6];

  float* Y = (float*)d_out;
  float* Hlast = Y + (size_t)Bb * Tt * Oo;

  char* ws = (char*)d_ws;
  unsigned short* Ubf = (unsigned short*)(ws + OFF_UBF);
  unsigned short* hb0 = (unsigned short*)(ws + OFF_HB0);
  unsigned short* Vbf = (unsigned short*)(ws + OFF_VBF);
  unsigned int*   cnt = (unsigned int*)(ws + OFF_CNT);
  unsigned short* hsb = (unsigned short*)(ws + OFF_HS);

  rnn_init<<<4096, 256, 0, stream>>>(h0, U, V, Ubf, hb0, Vbf, cnt);
  rnn_scan<<<GROUPS * SLICES, 128, 0, stream>>>(X, W, U_b, Ubf, hb0, hsb, cnt,
                                                Hlast);
  rnn_out<<<(Bb * Tt) / ROWS, 128, 0, stream>>>(hsb, Vbf, V_b, Y);
}
